// Head_1236950581856
// MI455X (gfx1250) — matrix operand order repaired and checked by reading, compiled, not run on hardware
//
#include <hip/hip_runtime.h>
#include <hip/hip_bf16.h>

// ---------------------------------------------------------------------------
// Single attention head, B=8 T=2048 C=1024 HS=64, flash-attention with
// v_wmma_f32_16x16x32_bf16 on gfx1250 (wave32).
// ---------------------------------------------------------------------------

typedef __bf16 bf16_t;
typedef __attribute__((ext_vector_type(8)))  __bf16 v8bf;
typedef __attribute__((ext_vector_type(16))) __bf16 v16bf;
typedef __attribute__((ext_vector_type(8)))  float  v8f;

#define BATCH 8
#define TLEN  2048
#define CDIM  1024
#define HS    64

__device__ __forceinline__ v16bf cat8(v8bf lo, v8bf hi) {
  return __builtin_shufflevector(lo, hi, 0,1,2,3,4,5,6,7,8,9,10,11,12,13,14,15);
}

__device__ __forceinline__ v8f wmma_bf16(v16bf a, v16bf b, v8f c) {
  return __builtin_amdgcn_wmma_f32_16x16x32_bf16(false, a, false, b, (short)0, c,
                                                 false, false);
}

// ---- kernel 0a: x (f32) -> xb (bf16), 8 elements / thread -----------------
__global__ void __launch_bounds__(256) cvt_x_kernel(const float4* __restrict__ x,
                                                    v8bf* __restrict__ xb, int n8) {
  int i = blockIdx.x * 256 + threadIdx.x;
  if (i >= n8) return;
  float4 a = x[2 * i];
  float4 b = x[2 * i + 1];
  v8bf o;
  o[0] = (bf16_t)a.x; o[1] = (bf16_t)a.y; o[2] = (bf16_t)a.z; o[3] = (bf16_t)a.w;
  o[4] = (bf16_t)b.x; o[5] = (bf16_t)b.y; o[6] = (bf16_t)b.z; o[7] = (bf16_t)b.w;
  xb[i] = o;
}

// ---- kernel 0b: W[C,HS] f32 -> Wt[3][HS][C] bf16 (transpose+convert) ------
// The softmax scale C**-0.5 = 2^-5 is folded into Wq here: multiplying by a
// power of two is exact (exponent shift), so this matches scaling att later.
__global__ void __launch_bounds__(256) cvt_w_kernel(const float* __restrict__ Wq,
                                                    const float* __restrict__ Wk,
                                                    const float* __restrict__ Wv,
                                                    bf16_t* __restrict__ Wt) {
  int i = blockIdx.x * 256 + threadIdx.x;      // 3*64*1024 = 196608 threads
  int mat = i >> 16;                            // /65536
  int rem = i & 65535;
  int h   = rem >> 10;                          // /1024
  int c   = rem & 1023;
  const float* W = (mat == 0) ? Wq : ((mat == 1) ? Wk : Wv);
  const float s = (mat == 0) ? 0.03125f : 1.0f;
  Wt[i] = (bf16_t)(W[c * HS + h] * s);
}

// ---- kernel 1: projections. One wave per 16-row x 64-col tile -------------
// Q,K stored [B*T][HS] bf16 row-major; V stored transposed Vt[B][HS][T] bf16.
// Two-stage software pipeline, unroll disabled so the double-buffered
// fragment registers survive register allocation.
__global__ void __launch_bounds__(256) proj_kernel(const bf16_t* __restrict__ xb,
                                                   const bf16_t* __restrict__ Wt,
                                                   bf16_t* __restrict__ Q,
                                                   bf16_t* __restrict__ K,
                                                   bf16_t* __restrict__ Vt) {
  const int lane = threadIdx.x & 31;
  const int wid  = blockIdx.x * 8 + (threadIdx.x >> 5);  // 3072 waves
  const int mat  = wid >> 10;                  // 0=Q 1=K 2=V
  const int rowT = wid & 1023;
  const int gr0  = rowT * 16;                  // global row (b*T+t)
  const int m    = lane & 15;
  const int half = lane >> 4;

  v8f acc0 = {}, acc1 = {}, acc2 = {}, acc3 = {};
  const bf16_t* arow = xb + (size_t)(gr0 + m) * CDIM + half * 16;
  const bf16_t* brow = Wt + (size_t)(mat * HS + m) * CDIM + half * 16;

  // prologue: fragments for kk = 0
  v16bf a  = cat8(*(const v8bf*)(arow), *(const v8bf*)(arow + 8));
  v16bf b0 = *(const v16bf*)(brow + (size_t)(0 * 16) * CDIM);
  v16bf b1 = *(const v16bf*)(brow + (size_t)(1 * 16) * CDIM);
  v16bf b2 = *(const v16bf*)(brow + (size_t)(2 * 16) * CDIM);
  v16bf b3 = *(const v16bf*)(brow + (size_t)(3 * 16) * CDIM);

#pragma unroll 1
  for (int kk = 32; kk < CDIM; kk += 32) {
    // issue next step's loads first (independent of current WMMAs)
    v16bf an  = cat8(*(const v8bf*)(arow + kk), *(const v8bf*)(arow + kk + 8));
    v16bf bn0 = *(const v16bf*)(brow + (size_t)(0 * 16) * CDIM + kk);
    v16bf bn1 = *(const v16bf*)(brow + (size_t)(1 * 16) * CDIM + kk);
    v16bf bn2 = *(const v16bf*)(brow + (size_t)(2 * 16) * CDIM + kk);
    v16bf bn3 = *(const v16bf*)(brow + (size_t)(3 * 16) * CDIM + kk);
    acc0 = wmma_bf16(a, b0, acc0);
    acc1 = wmma_bf16(a, b1, acc1);
    acc2 = wmma_bf16(a, b2, acc2);
    acc3 = wmma_bf16(a, b3, acc3);
    a = an; b0 = bn0; b1 = bn1; b2 = bn2; b3 = bn3;
  }
  acc0 = wmma_bf16(a, b0, acc0);
  acc1 = wmma_bf16(a, b1, acc1);
  acc2 = wmma_bf16(a, b2, acc2);
  acc3 = wmma_bf16(a, b3, acc3);

  if (mat == 2) {
#pragma unroll
    for (int r = 0; r < 8; ++r) {
      int gr = gr0 + r + half * 8;
      int bb = gr >> 11;                       // /T
      int t  = gr & (TLEN - 1);
      bf16_t* dv = Vt + (size_t)(bb * HS + m) * TLEN + t;
      dv[(size_t)(0 * 16) * TLEN] = (bf16_t)acc0[r];
      dv[(size_t)(1 * 16) * TLEN] = (bf16_t)acc1[r];
      dv[(size_t)(2 * 16) * TLEN] = (bf16_t)acc2[r];
      dv[(size_t)(3 * 16) * TLEN] = (bf16_t)acc3[r];
    }
  } else {
    bf16_t* dst = (mat == 0) ? Q : K;
#pragma unroll
    for (int r = 0; r < 8; ++r) {
      int gr = gr0 + r + half * 8;
      bf16_t* dq = dst + (size_t)gr * HS + m;
      dq[0]  = (bf16_t)acc0[r];
      dq[16] = (bf16_t)acc1[r];
      dq[32] = (bf16_t)acc2[r];
      dq[48] = (bf16_t)acc3[r];
    }
  }
}

// ---- kernel 2: causal flash attention. One wave per 16-query tile. --------
__global__ void __launch_bounds__(256) attn_kernel(const bf16_t* __restrict__ Q,
                                                   const bf16_t* __restrict__ K,
                                                   const bf16_t* __restrict__ Vt,
                                                   float* __restrict__ out) {
  __shared__ bf16_t plds[8][16 * 32];          // per-wave P staging (1 KB each)
  const int lane = threadIdx.x & 31;
  const int w    = threadIdx.x >> 5;
  const int wid  = blockIdx.x * 8 + w;         // 1024 query tiles
  const int b    = wid >> 7;                   // /128
  const int idx  = wid & 127;
  // pair light and heavy causal tiles on adjacent waves: work per wave-pair
  // is constant (~65 key blocks) instead of a 64:1 skew.
  const int qi   = (idx & 1) ? (127 - (idx >> 1)) : (idx >> 1);
  const int q0   = qi * 16;
  const int m    = lane & 15;
  const int half = lane >> 4;

  // Q A-fragments (hs 0..31, 32..63), loaded once (scale pre-folded into Wq)
  const bf16_t* qrow = Q + (size_t)(b * TLEN + q0 + m) * HS + half * 16;
  v16bf qa0 = cat8(*(const v8bf*)(qrow),      *(const v8bf*)(qrow + 8));
  v16bf qa1 = cat8(*(const v8bf*)(qrow + 32), *(const v8bf*)(qrow + 40));

  v8f o0 = {}, o1 = {}, o2 = {}, o3 = {};
  float mrow[8], lrow[8];
#pragma unroll
  for (int r = 0; r < 8; ++r) { mrow[r] = -__builtin_inff(); lrow[r] = 0.f; }

  bf16_t* pw = &plds[w][0];
  const int nkb = (q0 + 16 + 31) >> 5;         // 32-key blocks to visit
  const bf16_t* kbb = K + (size_t)(b * TLEN + m) * HS + half * 16;
  const bf16_t* vbb = Vt + (size_t)(b * HS) * TLEN + half * 16;

  // prologue: K fragments for key block 0
  v16bf kf00 = *(const v16bf*)(kbb);
  v16bf kf01 = *(const v16bf*)(kbb + 32);
  v16bf kf10 = *(const v16bf*)(kbb + (size_t)16 * HS);
  v16bf kf11 = *(const v16bf*)(kbb + (size_t)16 * HS + 32);

#pragma unroll 1
  for (int kb = 0; kb < nkb; ++kb) {
    const int k0 = kb * 32;

    // ---- S = Q K^T for 16 queries x 32 keys (two 16x16 D tiles) ----
    v8f s0 = {}, s1 = {};
    s0 = wmma_bf16(qa0, kf00, s0);
    s0 = wmma_bf16(qa1, kf01, s0);
    s1 = wmma_bf16(qa0, kf10, s1);
    s1 = wmma_bf16(qa1, kf11, s1);

    // V fragments for this block (independent of the softmax below)
    const bf16_t* vb = vbb + k0;
    v16bf vf0 = *(const v16bf*)(vb + (size_t)(0  + m) * TLEN);
    v16bf vf1 = *(const v16bf*)(vb + (size_t)(16 + m) * TLEN);
    v16bf vf2 = *(const v16bf*)(vb + (size_t)(32 + m) * TLEN);
    v16bf vf3 = *(const v16bf*)(vb + (size_t)(48 + m) * TLEN);

    // prefetch next key block's K fragments (clamped: stay in-bounds)
    const int kpre = (kb + 1 < nkb) ? (k0 + 32) : k0;
    {
      const bf16_t* kn = kbb + (size_t)kpre * HS;
      kf00 = *(const v16bf*)(kn);
      kf01 = *(const v16bf*)(kn + 32);
      kf10 = *(const v16bf*)(kn + (size_t)16 * HS);
      kf11 = *(const v16bf*)(kn + (size_t)16 * HS + 32);
    }

    // ---- online softmax (row stats live in the 16-lane half owning the row)
    float corr[8];
#pragma unroll
    for (int r = 0; r < 8; ++r) {
      const int qabs = q0 + r + half * 8;
      float v0 = s0[r];
      float v1 = s1[r];
      if (k0 + m      > qabs) v0 = -__builtin_inff();   // causal mask
      if (k0 + 16 + m > qabs) v1 = -__builtin_inff();
      float mx = fmaxf(v0, v1);
      mx = fmaxf(mx, __shfl_xor(mx, 1, 32));
      mx = fmaxf(mx, __shfl_xor(mx, 2, 32));
      mx = fmaxf(mx, __shfl_xor(mx, 4, 32));
      mx = fmaxf(mx, __shfl_xor(mx, 8, 32));
      const float mnew = fmaxf(mrow[r], mx);
      const float c  = __expf(mrow[r] - mnew);
      const float e0 = __expf(v0 - mnew);
      const float e1 = __expf(v1 - mnew);
      float sum = e0 + e1;
      sum += __shfl_xor(sum, 1, 32);
      sum += __shfl_xor(sum, 2, 32);
      sum += __shfl_xor(sum, 4, 32);
      sum += __shfl_xor(sum, 8, 32);
      lrow[r] = lrow[r] * c + sum;
      mrow[r] = mnew;
      corr[r] = c;
      pw[(r + half * 8) * 32 + m]      = (bf16_t)e0;   // stage P in LDS
      pw[(r + half * 8) * 32 + 16 + m] = (bf16_t)e1;
    }
    asm volatile("s_wait_dscnt 0" ::: "memory");        // LDS writes visible

    // re-read P in A-fragment layout (16x32 bf16)
    const bf16_t* pr = pw + m * 32 + half * 16;
    v16bf pa = cat8(*(const v8bf*)pr, *(const v8bf*)(pr + 8));

    // ---- O = diag(corr)*O + P @ V ----
#pragma unroll
    for (int r = 0; r < 8; ++r) {
      o0[r] *= corr[r]; o1[r] *= corr[r]; o2[r] *= corr[r]; o3[r] *= corr[r];
    }
    o0 = wmma_bf16(pa, vf0, o0);
    o1 = wmma_bf16(pa, vf1, o1);
    o2 = wmma_bf16(pa, vf2, o2);
    o3 = wmma_bf16(pa, vf3, o3);
  }

  // ---- finalize: divide rows by l, store fp32 ----
  float* orow = out + (size_t)(b * TLEN + q0 + half * 8) * HS + m;
#pragma unroll
  for (int r = 0; r < 8; ++r) {
    const float inv = 1.0f / lrow[r];
    orow[(size_t)r * HS + 0]  = o0[r] * inv;
    orow[(size_t)r * HS + 16] = o1[r] * inv;
    orow[(size_t)r * HS + 32] = o2[r] * inv;
    orow[(size_t)r * HS + 48] = o3[r] * inv;
  }
}

// ---------------------------------------------------------------------------
extern "C" void kernel_launch(void* const* d_in, const int* in_sizes, int n_in,
                              void* d_out, int out_size, void* d_ws, size_t ws_size,
                              hipStream_t stream) {
  const float* x  = (const float*)d_in[0];
  const float* Wq = (const float*)d_in[1];
  const float* Wk = (const float*)d_in[2];
  const float* Wv = (const float*)d_in[3];
  float* out = (float*)d_out;

  char* ws = (char*)d_ws;
  // workspace layout (bytes)
  bf16_t* xb = (bf16_t*)(ws);                                   // 33,554,432
  bf16_t* Wt = (bf16_t*)(ws + 33554432);                        //    393,216
  bf16_t* Qb = (bf16_t*)(ws + 33554432 + 393216);               //  2,097,152
  bf16_t* Kb = (bf16_t*)(ws + 33554432 + 393216 + 2097152);     //  2,097,152
  bf16_t* Vt = (bf16_t*)(ws + 33554432 + 393216 + 2 * 2097152); //  2,097,152

  const int n8 = (BATCH * TLEN * CDIM) / 8;    // 2,097,152 v8bf stores
  cvt_x_kernel<<<n8 / 256, 256, 0, stream>>>((const float4*)x, (v8bf*)xb, n8);
  cvt_w_kernel<<<768, 256, 0, stream>>>(Wq, Wk, Wv, Wt);
  proj_kernel<<<384, 256, 0, stream>>>(xb, Wt, Qb, Kb, Vt);
  attn_kernel<<<128, 256, 0, stream>>>(Qb, Kb, Vt, out);
}
